// Euc_branch_57750130262067
// MI455X (gfx1250) — matrix operand order repaired and checked by reading, not compiled, not run on hardware
//
#include <hip/hip_runtime.h>

typedef __attribute__((ext_vector_type(16))) _Float16 v16h;
typedef __attribute__((ext_vector_type(8)))  _Float16 v8h;
typedef __attribute__((ext_vector_type(8)))  float    v8f;
typedef __attribute__((ext_vector_type(4)))  float    v4f;

__device__ __forceinline__ int iclampi(int x, int lo, int hi) {
    return x < lo ? lo : (x > hi ? hi : x);
}

// ---------------------------------------------------------------------------
// Weight repack: f32 OIDHW (io=0) or IODHW (io=1) -> f16 [tap][Cout][CinPad]
// (CinPad zero-fills channels >= Cin, used to pad the 3-ch input conv to 32)
// ---------------------------------------------------------------------------
__global__ void repack_w_k(const float* __restrict__ w, _Float16* __restrict__ wp,
                           int Cout, int Cin, int CinPad, int taps, int io)
{
    long t = (long)blockIdx.x * blockDim.x + threadIdx.x;
    long total = (long)taps * Cout * CinPad;
    if (t >= total) return;
    int ci = (int)(t % CinPad);
    long r = t / CinPad;
    int co = (int)(r % Cout);
    int tap = (int)(r / Cout);
    float v = 0.f;
    if (ci < Cin) {
        long src = io ? (((long)ci * Cout + co) * taps + tap)
                      : (((long)co * Cin + ci) * taps + tap);
        v = w[src];
    }
    wp[t] = (_Float16)v;
}

__global__ void zero_f32_k(float* __restrict__ p, long n)
{
    long t = (long)blockIdx.x * blockDim.x + threadIdx.x;
    if (t < n) p[t] = 0.f;
}

// ---------------------------------------------------------------------------
// Scatter-mean voxelize: atomic f32 sums [vox][Cstride] + counts [vox]
// ---------------------------------------------------------------------------
__global__ void scatter_k(const float* __restrict__ pc, const float* __restrict__ sf,
                          const _Float16* __restrict__ sh, int Csrc, int Cstride,
                          float inv, int Rs, float* __restrict__ sums,
                          float* __restrict__ cnt, long N)
{
    long t = (long)blockIdx.x * blockDim.x + threadIdx.x;
    if (t >= N * Csrc) return;
    int c = (int)(t % Csrc);
    long n = t / Csrc;
    int ix = iclampi((int)floorf(pc[n * 3 + 0] * inv), 0, Rs - 1);
    int iy = iclampi((int)floorf(pc[n * 3 + 1] * inv), 0, Rs - 1);
    int iz = iclampi((int)floorf(pc[n * 3 + 2] * inv), 0, Rs - 1);
    long v = ((long)ix * Rs + iy) * Rs + iz;
    float val = sf ? sf[t] : (float)sh[t];
    atomicAdd(&sums[v * Cstride + c], val);
    if (c == 0) atomicAdd(&cnt[v], 1.0f);
}

__global__ void finalize_k(const float* __restrict__ sums, const float* __restrict__ cnt,
                           _Float16* __restrict__ g, long V, int Csrc, int Cstride)
{
    long t = (long)blockIdx.x * blockDim.x + threadIdx.x;
    if (t >= V * Cstride) return;
    int c = (int)(t % Cstride);
    long v = t / Cstride;
    float val = (c < Csrc) ? sums[t] / fmaxf(cnt[v], 1.f) : 0.f;
    g[t] = (_Float16)val;
}

// ---------------------------------------------------------------------------
// Trilinear devoxelize (+ optional fused add of point-branch features)
// ---------------------------------------------------------------------------
__global__ void devox_k(const _Float16* __restrict__ grid, int Rs, int C,
                        const float* __restrict__ pc, const _Float16* __restrict__ zpt,
                        _Float16* __restrict__ z, float inv, long N)
{
    long t = (long)blockIdx.x * blockDim.x + threadIdx.x;
    if (t >= N * C) return;
    int c = (int)(t % C);
    long n = t / C;
    float cx = pc[n * 3 + 0] * inv, cy = pc[n * 3 + 1] * inv, cz = pc[n * 3 + 2] * inv;
    float fx = floorf(cx), fy = floorf(cy), fz = floorf(cz);
    float ax = cx - fx, ay = cy - fy, az = cz - fz;
    int x0 = (int)fx, y0 = (int)fy, zz0 = (int)fz;
    float acc = 0.f;
    for (int dx = 0; dx < 2; dx++)
        for (int dy = 0; dy < 2; dy++)
            for (int dz = 0; dz < 2; dz++) {
                int xi = iclampi(x0 + dx, 0, Rs - 1);
                int yi = iclampi(y0 + dy, 0, Rs - 1);
                int zi = iclampi(zz0 + dz, 0, Rs - 1);
                float w = (dx ? ax : 1.f - ax) * (dy ? ay : 1.f - ay) * (dz ? az : 1.f - az);
                acc += w * (float)grid[(((long)xi * Rs + yi) * Rs + zi) * C + c];
            }
    if (zpt) acc += (float)zpt[t];
    z[t] = (_Float16)acc;
}

// skip-copy into channel slice of a concat buffer
__global__ void copych_k(const _Float16* __restrict__ src, int srcStride, int C,
                         _Float16* __restrict__ dst, int dstStride, int dstOff, long V)
{
    long t = (long)blockIdx.x * blockDim.x + threadIdx.x;
    if (t >= V * C) return;
    int c = (int)(t % C);
    long v = t / C;
    dst[v * dstStride + dstOff + c] = src[v * srcStride + c];
}

// channels-last f16 grid -> NCDHW f32 output slice
__global__ void export_k(const _Float16* __restrict__ g, int C, long V, float* __restrict__ dst)
{
    long t = (long)blockIdx.x * blockDim.x + threadIdx.x;
    if (t >= (long)C * V) return;
    long v = t % V;
    long c = t / V;
    dst[t] = (float)g[v * C + c];
}

// final 32 -> 20 linear head (tiny, bandwidth-trivial)
__global__ void outlin_k(const _Float16* __restrict__ z, const float* __restrict__ w,
                         const float* __restrict__ b, float* __restrict__ dst, long N)
{
    long t = (long)blockIdx.x * blockDim.x + threadIdx.x;
    if (t >= N * 20) return;
    int o = (int)(t % 20);
    long n = t / 20;
    float a = b[o];
    for (int i = 0; i < 32; i++) a += (float)z[n * 32 + i] * w[i * 20 + o];
    dst[t] = a;
}

// ---------------------------------------------------------------------------
// Implicit-GEMM conv via V_WMMA_F32_16X16X32_F16 (wave32).
//   MODE 0: 3x3x3 stride1 SAME     MODE 1: 2x2x2 stride2 VALID (downsample)
//   MODE 2: 1x1x1 (shortcuts & point MLPs)   MODE 3: 2x stride-2 deconv
// Register-blocked along N: one wave owns a 16(cout) x (16*NT)(voxel) f32
// tile (NT v8f accumulators). Per K-step the A (weight) registers are loaded
// once and reused for NT WMMAs -> 4x less A traffic, amortized tap math.
// A = weights [tap][Cout][Cin] f16 ; B = channels-last activations.
// Epilogue fuses bias (point MLP), BN affine, residual add, ReLU and writes
// each lane's 8 consecutive f16 results as ONE 16B vector store (the 8
// channels per lane are contiguous and 16B aligned in channels-last layout);
// gamma/beta/bias are explicit float4 loads hoisted out of the subtile loop,
// the residual row is a single v8h load.
// Deconv tiles output voxels per parity class so the kernel tap (A matrix)
// stays wave-uniform, making transposed conv a single-tap GEMM (NT=1).
// Wave-uniform ids go through readfirstlane so tiling math + the early-exit
// guard stay on the SALU path and EXEC is provably all-ones at the WMMAs.
// ---------------------------------------------------------------------------
template <int MODE, int NT>
__global__ __launch_bounds__(128)
void conv_wmma_k(const _Float16* __restrict__ in, const _Float16* __restrict__ wp,
                 const float* __restrict__ gamma, const float* __restrict__ beta,
                 const float* __restrict__ bias, const _Float16* __restrict__ addsrc,
                 _Float16* __restrict__ out,
                 int D, int H, int W, int Cin, int Cout,
                 int inStride, int outStride, int addStride, int coutOff, int relu)
{
    const int lane = threadIdx.x & 31;
    const int col = lane & 15;
    const int half = lane >> 4;
    const int nVox = D * H * W;

    int tiles16, groups, inVox = 0, tilesPerClass = 0;
    if (MODE == 3) {
        inVox = nVox >> 3;
        tilesPerClass = (inVox + 15) >> 4;
        tiles16 = tilesPerClass << 3;
        groups = tiles16;                    // NT must be 1 for MODE 3
    } else {
        tiles16 = (nVox + 15) >> 4;
        groups = (tiles16 + NT - 1) / NT;
    }
    const int nCoutBlk = Cout >> 4;
    const int wlocal = __builtin_amdgcn_readfirstlane((int)(threadIdx.x >> 5));
    const int wave = (int)blockIdx.x * (int)(blockDim.x >> 5) + wlocal;
    if (wave >= groups * nCoutBlk) return;   // scalar, wave-uniform exit
    const int coutBlk = wave / groups;
    const int grp = wave - coutBlk * groups;

    // per-subtile column info
    int outVox[NT], od[NT], oh[NT], ow[NT];
    bool nv[NT];
    int p = 0, q = 0;
    if (MODE == 3) {
        p = grp / tilesPerClass;
        const int qt = grp - p * tilesPerClass;
        q = qt * 16 + col;
        nv[0] = q < inVox;
        const int hw2 = (H >> 1) * (W >> 1);
        const int qq = nv[0] ? q : 0;
        const int qd = qq / hw2;
        int rr = qq - qd * hw2;
        const int qh = rr / (W >> 1);
        const int qw = rr - qh * (W >> 1);
        od[0] = 2 * qd + ((p >> 2) & 1);
        oh[0] = 2 * qh + ((p >> 1) & 1);
        ow[0] = 2 * qw + (p & 1);
        outVox[0] = (od[0] * H + oh[0]) * W + ow[0];
    } else {
#pragma unroll
        for (int j = 0; j < NT; j++) {
            const int vt16 = grp * NT + j;
            const int v = vt16 * 16 + col;
            nv[j] = v < nVox;
            const int vv = nv[j] ? v : 0;
            const int hw = H * W;
            od[j] = vv / hw;
            int rr = vv - od[j] * hw;
            oh[j] = rr / W;
            ow[j] = rr - oh[j] * W;
            outVox[j] = v;
        }
    }

    const int nTaps = (MODE == 0) ? 27 : (MODE == 1) ? 8 : 1;
    const size_t tapStride = (size_t)Cout * Cin;

    v8f acc[NT];
#pragma unroll
    for (int j = 0; j < NT; j++)
#pragma unroll
        for (int i = 0; i < 8; i++) acc[j][i] = 0.f;

    const _Float16* aRow = wp + (size_t)(coutBlk * 16 + col) * Cin +
                           ((MODE == 3) ? (size_t)p * tapStride : (size_t)0);

    for (int t = 0; t < nTaps; ++t) {
        size_t bOff[NT];
        bool sv[NT];
#pragma unroll
        for (int j = 0; j < NT; j++) {
            long srcVox;
            if (MODE == 0) {
                const int kd = t / 9, kh = (t / 3) % 3, kw = t % 3;
                const int id = od[j] + kd - 1, ih = oh[j] + kh - 1, iw = ow[j] + kw - 1;
                sv[j] = nv[j] && ((unsigned)id < (unsigned)D) &&
                        ((unsigned)ih < (unsigned)H) && ((unsigned)iw < (unsigned)W);
                srcVox = sv[j] ? (((long)id * H + ih) * W + iw) : 0;
            } else if (MODE == 1) {
                const int kd = (t >> 2) & 1, kh = (t >> 1) & 1, kw = t & 1;
                const int id = 2 * od[j] + kd, ih = 2 * oh[j] + kh, iw = 2 * ow[j] + kw;
                sv[j] = nv[j];
                srcVox = ((long)id * (2 * H) + ih) * (2 * W) + iw;
            } else if (MODE == 2) {
                sv[j] = nv[j];
                srcVox = outVox[j];
            } else {
                sv[j] = nv[j];
                srcVox = q;
            }
            bOff[j] = (size_t)srcVox * inStride;
        }
        const _Float16* aBase = aRow + ((MODE == 3) ? (size_t)0 : (size_t)t * tapStride);
        for (int cb = 0; cb < Cin; cb += 32) {
            // A (16x32 f16): lanes 0-15 K=0..15 ; lanes 16-31 K=16..31
            v16h A;
            {
                A = *(const v16h*)(aBase + cb + half * 16);
            }
            // B (32x16 f16): lane -> column voxel, 16 contiguous channels
#pragma unroll
            for (int j = 0; j < NT; j++) {
                v16h B;
#pragma unroll
                for (int i = 0; i < 16; i++) B[i] = (_Float16)0.f;
                if (sv[j]) B = *(const v16h*)(in + bOff[j] + cb + (half << 4));
                acc[j] = __builtin_amdgcn_wmma_f32_16x16x32_f16(false, A, false, B,
                                                                (short)0, acc[j], false, false);
            }
        }
    }

    // ---- epilogue: one 16B store per lane per subtile ----
    const int mBase = coutBlk * 16 + half * 8;   // C/D layout: VGPR r -> M=r+8*half
    const v4f g0 = *(const v4f*)(gamma + mBase);
    const v4f g1 = *(const v4f*)(gamma + mBase + 4);
    const v4f b0 = *(const v4f*)(beta + mBase);
    const v4f b1 = *(const v4f*)(beta + mBase + 4);
    v4f s0, s1;
    if (bias) { s0 = *(const v4f*)(bias + mBase); s1 = *(const v4f*)(bias + mBase + 4); }
#pragma unroll
    for (int j = 0; j < NT; j++) {
        if (!nv[j]) continue;
        v8h av;
        if (addsrc) av = *(const v8h*)(addsrc + (size_t)outVox[j] * addStride + mBase);
        v8h hv;
#pragma unroll
        for (int r = 0; r < 8; r++) {
            float v = acc[j][r];
            if (bias) v += (r < 4) ? s0[r] : s1[r - 4];
            v = v * ((r < 4) ? g0[r] : g1[r - 4]) + ((r < 4) ? b0[r] : b1[r - 4]);
            if (addsrc) v += (float)av[r];
            if (relu) v = fmaxf(v, 0.f);
            hv[r] = (_Float16)v;
        }
        *(v8h*)(out + (size_t)outVox[j] * outStride + coutOff + mBase) = hv;
    }
}

// ---------------------------------------------------------------------------
extern "C" void kernel_launch(void* const* d_in, const int* in_sizes, int n_in,
                              void* d_out, int out_size, void* d_ws, size_t ws_size,
                              hipStream_t stream)
{
    (void)n_in; (void)out_size; (void)ws_size;
    const float* pf = (const float*)d_in[0];
    const float* pc = (const float*)d_in[1];
    const long N = in_sizes[0] / 3;

    // ---- parse params (insertion order of the reference dicts) ----
    int li = 2;
    auto P = [&](void) -> const float* { return (const float*)d_in[li++]; };
    struct CBR { const float *w, *g, *b; };
    auto RC = [&](void) -> CBR { CBR c; c.w = P(); c.g = P(); c.b = P(); return c; };

    CBR ic0 = RC(), ic1 = RC();
    CBR sd[6], s1c[6], s1d[6], s2c[6];
    for (int s = 0; s < 6; s++) { sd[s] = RC(); s1c[s] = RC(); s1d[s] = RC(); s2c[s] = RC(); }
    CBR ude[6], u1c[6], u1d[6], u2c[6];
    for (int u = 0; u < 6; u++) { ude[u] = RC(); u1c[u] = RC(); u1d[u] = RC(); u2c[u] = RC(); }
    struct PT { const float *w, *bias, *g, *b; };
    PT pt[4];
    for (int k = 0; k < 4; k++) { pt[k].w = P(); pt[k].bias = P(); pt[k].g = P(); pt[k].b = P(); }
    const float* outW = P();
    const float* outB = P();

    static const int eIn[6] = {32, 64, 96, 128, 160, 192}, eOut[6] = {64, 96, 128, 160, 192, 224};
    static const int dIn[6] = {224, 192, 160, 128, 96, 64}, dOut[6] = {192, 160, 128, 96, 64, 32},
                     dSk[6] = {192, 160, 128, 96, 64, 32};
    static const int ptI[4] = {32, 224, 160, 96}, ptO[4] = {224, 160, 96, 32};

    // ---- workspace bump allocator ----
    char* base = (char*)d_ws;
    size_t off = 0;
    auto allocB = [&](size_t bytes) -> char* {
        char* r = base + off;
        off += (bytes + 255) & ~(size_t)255;
        return r;
    };
    auto aH = [&](size_t n) -> _Float16* { return (_Float16*)allocB(n * sizeof(_Float16)); };
    auto aF = [&](size_t n) -> float* { return (float*)allocB(n * sizeof(float)); };
    auto GB = [&](long n) -> dim3 { return dim3((unsigned)((n + 255) / 256)); };

    auto repack = [&](const float* w, _Float16* wpk, int Co, int Ci, int CiP, int taps, int io) {
        long tot = (long)taps * Co * CiP;
        repack_w_k<<<GB(tot), 256, 0, stream>>>(w, wpk, Co, Ci, CiP, taps, io);
    };

    // ---- repack all weights to f16 GEMM layout [tap][Cout][Cin] ----
    _Float16* wIC0 = aH((size_t)27 * 32 * 32); repack(ic0.w, wIC0, 32, 3, 32, 27, 0);
    _Float16* wIC1 = aH((size_t)27 * 32 * 32); repack(ic1.w, wIC1, 32, 32, 32, 27, 0);
    _Float16 *wSd[6], *wS1[6], *wS1d[6], *wS2[6];
    for (int s = 0; s < 6; s++) {
        int i = eIn[s], o = eOut[s];
        wSd[s]  = aH((size_t)8 * i * i);  repack(sd[s].w,  wSd[s],  i, i, i, 8, 0);
        wS1[s]  = aH((size_t)27 * o * i); repack(s1c[s].w, wS1[s],  o, i, i, 27, 0);
        wS1d[s] = aH((size_t)o * i);      repack(s1d[s].w, wS1d[s], o, i, i, 1, 0);
        wS2[s]  = aH((size_t)27 * o * o); repack(s2c[s].w, wS2[s],  o, o, o, 27, 0);
    }
    _Float16 *wUde[6], *wU1[6], *wU1d[6], *wU2[6];
    for (int u = 0; u < 6; u++) {
        int i = dIn[u], o = dOut[u], ct = dOut[u] + dSk[u];
        wUde[u] = aH((size_t)8 * o * i);   repack(ude[u].w, wUde[u], o, i, i, 8, 1);
        wU1[u]  = aH((size_t)27 * o * ct); repack(u1c[u].w, wU1[u],  o, ct, ct, 27, 0);
        wU1d[u] = aH((size_t)o * ct);      repack(u1d[u].w, wU1d[u], o, ct, ct, 1, 0);
        wU2[u]  = aH((size_t)27 * o * o);  repack(u2c[u].w, wU2[u],  o, o, o, 27, 0);
    }
    _Float16* wPt[4];
    for (int k = 0; k < 4; k++) {
        wPt[k] = aH((size_t)ptO[k] * ptI[k]);
        repack(pt[k].w, wPt[k], ptO[k], ptI[k], ptI[k], 1, 1);
    }

    // ---- activation buffers (f16 channels-last) ----
    const long V0 = 64L * 64 * 64;
    _Float16* gPad = aH(V0 * 32);
    _Float16* gT   = aH(V0 * 32);
    _Float16* gX0  = aH(V0 * 32);
    _Float16* gXin = aH(V0 * 32);
    _Float16* encB[6];
    for (int s = 0; s < 6; s++) { long g = 64L >> (s + 1); encB[s] = aH(g * g * g * (long)eOut[s]); }
    _Float16* yB[6];
    for (int u = 0; u < 6; u++) { long g = 2L << u; yB[u] = aH(g * g * g * (long)dOut[u]); }
    _Float16* gY6 = aH(224);
    _Float16* gZ2 = aH(64L * 160);
    _Float16* gZ3 = aH(4096L * 96);
    _Float16* tmp1 = aH((size_t)V0 * 64);  // concat / downsample temp
    _Float16* tmp2 = aH((size_t)V0 * 32);  // residual-shortcut temp
    _Float16* tmp3 = aH((size_t)V0 * 32);  // r1 output temp
    _Float16* z0 = aH((size_t)N * 32);
    _Float16* z1 = aH((size_t)N * 224);
    _Float16* z2 = aH((size_t)N * 160);
    _Float16* z3 = aH((size_t)N * 96);
    _Float16* z4 = aH((size_t)N * 32);
    _Float16* zpt = aH((size_t)N * 224);
    float* sums = aF(V0 * 32);
    float* cnt = aF(V0);

    auto zeroF = [&](float* pz, long n) { zero_f32_k<<<GB(n), 256, 0, stream>>>(pz, n); };

    auto vox = [&](const float* sf, const _Float16* sh, int Csrc, int Cstr, float inv, int Rs,
                   _Float16* gdst) {
        long V = (long)Rs * Rs * Rs;
        zeroF(sums, V * Cstr);
        zeroF(cnt, V);
        scatter_k<<<GB(N * (long)Csrc), 256, 0, stream>>>(pc, sf, sh, Csrc, Cstr, inv, Rs, sums, cnt, N);
        finalize_k<<<GB(V * Cstr), 256, 0, stream>>>(sums, cnt, gdst, V, Csrc, Cstr);
    };

    auto devox = [&](const _Float16* g, int Rs, int C, const _Float16* zp, _Float16* zdst, float inv) {
        devox_k<<<GB(N * (long)C), 256, 0, stream>>>(g, Rs, C, pc, zp, zdst, inv, N);
    };

    auto conv = [&](int mode, const _Float16* in, const _Float16* wpk, const float* g,
                    const float* b, const float* bias, const _Float16* add, _Float16* outp,
                    int D, int Hh, int Ww, int Cin, int Cout, int inStr, int outStr,
                    int addStr, int coutOff, int relu) {
        long nVox = (long)D * Hh * Ww;
        long groups;
        if (mode == 3) {
            groups = 8L * ((nVox / 8 + 15) / 16);
        } else {
            long tiles16 = (nVox + 15) / 16;
            groups = (tiles16 + 3) / 4;      // NT = 4
        }
        long waves = groups * (Cout / 16);
        unsigned blocks = (unsigned)((waves + 3) / 4);
        switch (mode) {
        case 0: conv_wmma_k<0, 4><<<blocks, 128, 0, stream>>>(in, wpk, g, b, bias, add, outp, D, Hh, Ww, Cin, Cout, inStr, outStr, addStr, coutOff, relu); break;
        case 1: conv_wmma_k<1, 4><<<blocks, 128, 0, stream>>>(in, wpk, g, b, bias, add, outp, D, Hh, Ww, Cin, Cout, inStr, outStr, addStr, coutOff, relu); break;
        case 2: conv_wmma_k<2, 4><<<blocks, 128, 0, stream>>>(in, wpk, g, b, bias, add, outp, D, Hh, Ww, Cin, Cout, inStr, outStr, addStr, coutOff, relu); break;
        default: conv_wmma_k<3, 1><<<blocks, 128, 0, stream>>>(in, wpk, g, b, bias, add, outp, D, Hh, Ww, Cin, Cout, inStr, outStr, addStr, coutOff, relu); break;
        }
    };

    // ================= forward pass =================
    // initial voxelize (3ch padded to 32) + two input CBR convs
    vox(pf, nullptr, 3, 32, 1.f, 64, gPad);
    conv(0, gPad, wIC0, ic0.g, ic0.b, nullptr, nullptr, gT, 64, 64, 64, 32, 32, 32, 32, 0, 0, 1);
    conv(0, gT, wIC1, ic1.g, ic1.b, nullptr, nullptr, gX0, 64, 64, 64, 32, 32, 32, 32, 0, 0, 1);
    devox(gX0, 64, 32, nullptr, z0, 1.f);
    vox(nullptr, z0, 32, 32, 1.f, 64, gXin);

    // encoder
    const _Float16* cur = gXin;
    int dim = 64;
    for (int s = 0; s < 6; s++) {
        const int i = eIn[s], o = eOut[s];
        const int G = dim >> 1;
        conv(1, cur, wSd[s], sd[s].g, sd[s].b, nullptr, nullptr, tmp1, G, G, G, i, i, i, i, 0, 0, 1);
        conv(2, tmp1, wS1d[s], s1d[s].g, s1d[s].b, nullptr, nullptr, tmp2, G, G, G, i, o, i, o, 0, 0, 0);
        conv(0, tmp1, wS1[s], s1c[s].g, s1c[s].b, nullptr, tmp2, tmp3, G, G, G, i, o, i, o, o, 0, 1);
        conv(0, tmp3, wS2[s], s2c[s].g, s2c[s].b, nullptr, tmp3, encB[s], G, G, G, o, o, o, o, o, 0, 1);
        cur = encB[s];
        dim = G;
    }

    // z1 = devox(x6) + pt0(z0)  ;  y = voxelize(z1) @ Rs=1
    conv(2, z0, wPt[0], pt[0].g, pt[0].b, pt[0].bias, nullptr, zpt, (int)N, 1, 1, 32, 224, 32, 224, 0, 0, 1);
    devox(encB[5], 1, 224, zpt, z1, 1.f / 64.f);
    vox(nullptr, z1, 224, 224, 1.f / 64.f, 1, gY6);

    auto doUp = [&](int u, const _Float16* inG, int gdim, const _Float16* skip) {
        const int i = dIn[u], o = dOut[u], sk = dSk[u], ct = o + sk;
        const int G = gdim * 2;
        const long Vg = (long)G * G * G;
        conv(3, inG, wUde[u], ude[u].g, ude[u].b, nullptr, nullptr, tmp1, G, G, G, i, o, i, ct, 0, 0, 1);
        copych_k<<<GB(Vg * (long)sk), 256, 0, stream>>>(skip, sk, sk, tmp1, ct, o, Vg);
        conv(2, tmp1, wU1d[u], u1d[u].g, u1d[u].b, nullptr, nullptr, tmp2, G, G, G, ct, o, ct, o, 0, 0, 0);
        conv(0, tmp1, wU1[u], u1c[u].g, u1c[u].b, nullptr, tmp2, tmp3, G, G, G, ct, o, ct, o, o, 0, 1);
        conv(0, tmp3, wU2[u], u2c[u].g, u2c[u].b, nullptr, tmp3, yB[u], G, G, G, o, o, o, o, o, 0, 1);
    };

    doUp(0, gY6, 1, encB[4]);
    doUp(1, yB[0], 2, encB[3]);
    conv(2, z1, wPt[1], pt[1].g, pt[1].b, pt[1].bias, nullptr, zpt, (int)N, 1, 1, 224, 160, 224, 160, 0, 0, 1);
    devox(yB[1], 4, 160, zpt, z2, 1.f / 16.f);
    vox(nullptr, z2, 160, 160, 1.f / 16.f, 4, gZ2);
    doUp(2, gZ2, 4, encB[2]);
    doUp(3, yB[2], 8, encB[1]);
    conv(2, z2, wPt[2], pt[2].g, pt[2].b, pt[2].bias, nullptr, zpt, (int)N, 1, 1, 160, 96, 160, 96, 0, 0, 1);
    devox(yB[3], 16, 96, zpt, z3, 0.25f);
    vox(nullptr, z3, 96, 96, 0.25f, 16, gZ3);
    doUp(4, gZ3, 16, encB[0]);
    doUp(5, yB[4], 32, gX0);
    conv(2, z3, wPt[3], pt[3].g, pt[3].b, pt[3].bias, nullptr, zpt, (int)N, 1, 1, 96, 32, 96, 32, 0, 0, 1);
    devox(yB[5], 64, 32, zpt, z4, 1.f);

    // ---- outputs: point logits + 7 NCDHW grids, concatenated flat ----
    float* o = (float*)d_out;
    outlin_k<<<GB(N * 20), 256, 0, stream>>>(z4, outW, outB, o, N);
    o += N * 20;
    auto expg = [&](const _Float16* g, int C, long V) {
        export_k<<<GB((long)C * V), 256, 0, stream>>>(g, C, V, o);
        o += (long)C * V;
    };
    expg(encB[5], 224, 1);
    expg(yB[0], 192, 8);
    expg(yB[1], 160, 64);
    expg(yB[2], 128, 512);
    expg(yB[3], 96, 4096);
    expg(yB[4], 64, 32768);
    expg(yB[5], 32, 262144);
}
